// Model18_9620726743231
// MI455X (gfx1250) — matrix operand order repaired and checked by reading, not compiled, not run on hardware
//
#include <hip/hip_runtime.h>
#include <math.h>

// Problem constants (from reference)
#define UNITS 30
#define NN    131072
#define BBATCH 1024
#define EE    1310720
#define NBN   65536
#define EBN   262144
#define NBB   16384

typedef __attribute__((ext_vector_type(16))) _Float16 v16h;
typedef __attribute__((ext_vector_type(8)))  float    v8f;

// ---------------------------------------------------------------------------
// float atomic max via the signed/unsigned int trick (init to -inf)
// ---------------------------------------------------------------------------
__device__ __forceinline__ void atomicMaxF(float* addr, float v) {
  if (v >= 0.0f) atomicMax((int*)addr, __float_as_int(v));
  else           atomicMin((unsigned int*)addr, __float_as_uint(v));
}

__global__ void fill_kernel(float* __restrict__ p, long n, float v) {
  long i = (long)blockIdx.x * blockDim.x + threadIdx.x;
  if (i < n) p[i] = v;
}

// ---------------------------------------------------------------------------
// Weight pre-pack: W[30,Kd] (f32, row-major) -> f16 in the contiguous-k WMMA B-operand
// lane layout:  out[((s2*2+t)*32 + lane)*16 + j] = W[col*Kd + kk]
//   col = t*16 + (lane&15),  kk = s2*32 + (lane>>4)*16 + j,  zero padded.
// ---------------------------------------------------------------------------
__global__ void pack_w_kernel(const float* __restrict__ W, int Kd, int ksteps,
                              _Float16* __restrict__ out)
{
  int idx = blockIdx.x * blockDim.x + threadIdx.x;
  int total = ksteps * 1024;
  if (idx >= total) return;
  int j    = idx & 15;
  int lane = (idx >> 4) & 31;
  int t    = (idx >> 9) & 1;
  int s2   = idx >> 10;
  int khalf = lane >> 4;
  int col   = t * 16 + (lane & 15);
  int kk    = s2 * 32 + khalf * 16 + j;
  float wv = (col < UNITS && kk < Kd) ? W[col * Kd + kk] : 0.0f;
  out[idx] = (_Float16)wv;
}

__global__ void pack_b_kernel(const float* __restrict__ b, float* __restrict__ out)
{
  int c = threadIdx.x;               // one wave
  if (c < 32) out[c] = (c < UNITS) ? b[c] : 0.0f;
}

// ---------------------------------------------------------------------------
// A-matrix pack helper: per lane, two contiguous 8-float chunks per k-step
// (16-bit A 16x32 WMMA layout).  x padding columns are guaranteed zero.
// ---------------------------------------------------------------------------
__device__ __forceinline__ v16h pack_A(const float* __restrict__ xr,
                                       int s2, int khalf)
{
  const float4* p0 = reinterpret_cast<const float4*>(xr + s2 * 32 + khalf * 16);
  const float4* p1 = reinterpret_cast<const float4*>(xr + s2 * 32 + khalf * 16 + 8);
  float4 f0 = p0[0], f1 = p0[1];       // K = s2*32 + khalf*16 + 0..7
  float4 g0 = p1[0], g1 = p1[1];       // K = s2*32 + khalf*16 + 8..15
  v16h a;
  a[0]  = (_Float16)f0.x; a[1]  = (_Float16)f0.y;
  a[2]  = (_Float16)f0.z; a[3]  = (_Float16)f0.w;
  a[4]  = (_Float16)f1.x; a[5]  = (_Float16)f1.y;
  a[6]  = (_Float16)f1.z; a[7]  = (_Float16)f1.w;
  a[8]  = (_Float16)g0.x; a[9]  = (_Float16)g0.y;
  a[10] = (_Float16)g0.z; a[11] = (_Float16)g0.w;
  a[12] = (_Float16)g1.x; a[13] = (_Float16)g1.y;
  a[14] = (_Float16)g1.z; a[15] = (_Float16)g1.w;
  return a;
}

// ---------------------------------------------------------------------------
// Fused q/k/v/skip projection: pack A once, 4 matrices x 2 tiles x KSTEPS WMMAs.
// One wave32 per 16 rows.
// ---------------------------------------------------------------------------
template <int KSTEPS>
__global__ void qkvs_wmma_kernel(
    const float* __restrict__ xin, int strideIn,
    const _Float16* __restrict__ B0, const float* __restrict__ bp0, float* __restrict__ y0,
    const _Float16* __restrict__ B1, const float* __restrict__ bp1, float* __restrict__ y1,
    const _Float16* __restrict__ B2, const float* __restrict__ bp2, float* __restrict__ y2,
    const _Float16* __restrict__ B3, const float* __restrict__ bp3, float* __restrict__ y3,
    int n)
{
  const int lane  = threadIdx.x & 31;
  const int wave  = threadIdx.x >> 5;
  const int nodeBase = (blockIdx.x * (blockDim.x >> 5) + wave) * 16;
  if (nodeBase >= n) return;               // wave-uniform; EXEC stays all-1s
  const int mrow  = lane & 15;
  const int khalf = lane >> 4;

  v16h A[KSTEPS];
  const float* xr = xin + (size_t)(nodeBase + mrow) * (size_t)strideIn;
#pragma unroll
  for (int s2 = 0; s2 < KSTEPS; ++s2) A[s2] = pack_A(xr, s2, khalf);

  const _Float16* Bs[4]   = { B0, B1, B2, B3 };
  const float*    bps[4]  = { bp0, bp1, bp2, bp3 };
  float*          ys[4]   = { y0, y1, y2, y3 };

#pragma unroll
  for (int mtx = 0; mtx < 4; ++mtx) {
    const _Float16* Bp = Bs[mtx];
    float* y = ys[mtx];
#pragma unroll
    for (int t = 0; t < 2; ++t) {
      const int col = t * 16 + mrow;
      v8f acc = {};
#pragma unroll
      for (int s2 = 0; s2 < KSTEPS; ++s2) {
        v16h Bm = *reinterpret_cast<const v16h*>(
            Bp + ((size_t)((s2 * 2 + t) * 32 + lane)) * 16);
        acc = __builtin_amdgcn_wmma_f32_16x16x32_f16(
            false, A[s2], false, Bm, (short)0, acc, false, false);
      }
      const float bvv = bps[mtx][col];   // padded: zero for col>=30
#pragma unroll
      for (int r = 0; r < 8; ++r) {
        int row = nodeBase + r + khalf * 8;  // C layout: M = r + 8*khalf
        y[(size_t)row * 32 + col] = acc[r] + bvv;
      }
    }
  }
}

// ---------------------------------------------------------------------------
// Single dense linear with WMMA (used for final1): y = act(x @ W^T + b)
// ---------------------------------------------------------------------------
template <int KSTEPS>
__global__ void linear_wmma_kernel(
    const float* __restrict__ xin, int strideIn,
    const _Float16* __restrict__ Bp, const float* __restrict__ biasPad,
    float* __restrict__ y, int n, int relu)
{
  const int lane  = threadIdx.x & 31;
  const int wave  = threadIdx.x >> 5;
  const int nodeBase = (blockIdx.x * (blockDim.x >> 5) + wave) * 16;
  if (nodeBase >= n) return;
  const int mrow  = lane & 15;
  const int khalf = lane >> 4;

  v16h A[KSTEPS];
  const float* xr = xin + (size_t)(nodeBase + mrow) * (size_t)strideIn;
#pragma unroll
  for (int s2 = 0; s2 < KSTEPS; ++s2) A[s2] = pack_A(xr, s2, khalf);

#pragma unroll
  for (int t = 0; t < 2; ++t) {
    const int col = t * 16 + mrow;
    v8f acc = {};
#pragma unroll
    for (int s2 = 0; s2 < KSTEPS; ++s2) {
      v16h Bm = *reinterpret_cast<const v16h*>(
          Bp + ((size_t)((s2 * 2 + t) * 32 + lane)) * 16);
      acc = __builtin_amdgcn_wmma_f32_16x16x32_f16(
          false, A[s2], false, Bm, (short)0, acc, false, false);
    }
    const float bvv = biasPad[col];
#pragma unroll
    for (int r = 0; r < 8; ++r) {
      int row = nodeBase + r + khalf * 8;
      float val = acc[r] + bvv;
      if (relu) val = fmaxf(val, 0.0f);
      y[(size_t)row * 32 + col] = val;
    }
  }
}

// ---------------------------------------------------------------------------
// x0 = relu(W_init @ [graph_features, income[batch]] + b)     [N,7] -> [N,32]
// ---------------------------------------------------------------------------
__global__ void init_kernel(const float* __restrict__ gf,
                            const float* __restrict__ inc,
                            const int*   __restrict__ batch,
                            const float* __restrict__ W,
                            const float* __restrict__ b,
                            float* __restrict__ out, int n)
{
  int i = blockIdx.x * blockDim.x + threadIdx.x;
  if (i >= n) return;
  float f[7];
#pragma unroll
  for (int j = 0; j < 5; ++j) f[j] = gf[(size_t)i * 5 + j];
  int bb = batch[i];
  f[5] = inc[(size_t)bb * 2 + 0];
  f[6] = inc[(size_t)bb * 2 + 1];
  float* y = out + (size_t)i * 32;
#pragma unroll
  for (int c = 0; c < UNITS; ++c) {
    float acc = b[c];
#pragma unroll
    for (int j = 0; j < 7; ++j) acc += W[c * 7 + j] * f[j];
    y[c] = fmaxf(acc, 0.0f);
  }
  y[30] = 0.0f; y[31] = 0.0f;
}

// ---------------------------------------------------------------------------
// Edge attention kernels (TransformerConv, heads=1).
// 32-wide vector dots: padding cols are zero so the result is unchanged.
// ---------------------------------------------------------------------------
__device__ __forceinline__ float dot32(const float* __restrict__ a,
                                       const float* __restrict__ b)
{
  const float4* a4 = reinterpret_cast<const float4*>(a);
  const float4* b4 = reinterpret_cast<const float4*>(b);
  float acc = 0.0f;
#pragma unroll
  for (int i = 0; i < 8; ++i) {
    float4 x = a4[i], y = b4[i];
    acc += x.x * y.x + x.y * y.y + x.z * y.z + x.w * y.w;
  }
  return acc;
}

__global__ void score_kernel(const float* __restrict__ q,
                             const float* __restrict__ k,
                             const int* __restrict__ src,
                             const int* __restrict__ dst,
                             int ne, float scale,
                             float* __restrict__ sc, float* __restrict__ m)
{
  int e = blockIdx.x * blockDim.x + threadIdx.x;
  if (e >= ne) return;
  int s = src[e], d = dst[e];
  float acc = dot32(q + (size_t)d * 32, k + (size_t)s * 32) * scale;
  sc[e] = acc;
  atomicMaxF(&m[d], acc);
}

__global__ void fixm_kernel(float* __restrict__ m, int n) {
  int i = blockIdx.x * blockDim.x + threadIdx.x;
  if (i >= n) return;
  float v = m[i];
  if (v == -__builtin_inff()) m[i] = 0.0f;   // where(isfinite(m), m, 0)
}

__global__ void exp_kernel(float* __restrict__ sc,
                           const float* __restrict__ m,
                           float* __restrict__ ssum,
                           const int* __restrict__ dst, int ne)
{
  int e = blockIdx.x * blockDim.x + threadIdx.x;
  if (e >= ne) return;
  int d = dst[e];
  float ex = expf(sc[e] - m[d]);
  sc[e] = ex;
  atomicAdd(&ssum[d], ex);
}

__global__ void agg_kernel(const float* __restrict__ sc,
                           const float* __restrict__ ssum,
                           const float* __restrict__ v,
                           const int* __restrict__ src,
                           const int* __restrict__ dst,
                           int ne, float* __restrict__ agg)
{
  int e = blockIdx.x * blockDim.x + threadIdx.x;
  if (e >= ne) return;
  int d = dst[e];
  float a = sc[e] / ssum[d];
  const float4* vr = reinterpret_cast<const float4*>(v + (size_t)src[e] * 32);
  float* ar = agg + (size_t)d * 32;
#pragma unroll
  for (int i = 0; i < 8; ++i) {
    float4 x = vr[i];
    int c = i * 4;
    if (c + 0 < UNITS) atomicAdd(&ar[c + 0], a * x.x);
    if (c + 1 < UNITS) atomicAdd(&ar[c + 1], a * x.y);
    if (c + 2 < UNITS) atomicAdd(&ar[c + 2], a * x.z);
    if (c + 3 < UNITS) atomicAdd(&ar[c + 3], a * x.w);
  }
}

// beta-gated skip: y = beta*r + (1-beta)*o,  beta = sigmoid(bW . [o,r,o-r])
__global__ void gate_kernel(const float* __restrict__ agg,
                            const float* __restrict__ sk,
                            const float* __restrict__ bw,
                            float* __restrict__ out, int n, int relu)
{
  int i = blockIdx.x * blockDim.x + threadIdx.x;
  if (i >= n) return;
  const float* o = agg + (size_t)i * 32;
  const float* r = sk + (size_t)i * 32;
  float g = 0.0f;
#pragma unroll
  for (int c = 0; c < UNITS; ++c)
    g += bw[c] * o[c] + bw[UNITS + c] * r[c] + bw[2 * UNITS + c] * (o[c] - r[c]);
  float beta = 1.0f / (1.0f + expf(-g));
  float* y = out + (size_t)i * 32;
#pragma unroll
  for (int c = 0; c < UNITS; ++c) {
    float val = beta * r[c] + (1.0f - beta) * o[c];
    if (relu) val = fmaxf(val, 0.0f);
    y[c] = val;
  }
  y[30] = 0.0f; y[31] = 0.0f;
}

// ---------------------------------------------------------------------------
// Bonus path + readout kernels
// ---------------------------------------------------------------------------
__global__ void gather_kernel(const float* __restrict__ x,
                              const int* __restrict__ idx,
                              float* __restrict__ out, int n)
{
  int i = blockIdx.x * blockDim.x + threadIdx.x;
  if (i >= n) return;
  const float4* xr = reinterpret_cast<const float4*>(x + (size_t)idx[i] * 32);
  float4* y = reinterpret_cast<float4*>(out + (size_t)i * 32);
#pragma unroll
  for (int c = 0; c < 8; ++c) y[c] = xr[c];
}

__global__ void segpool_kernel(const float* __restrict__ x,
                               const int* __restrict__ seg,
                               float* __restrict__ pool, int n)
{
  int i = blockIdx.x * blockDim.x + threadIdx.x;
  if (i >= n) return;
  float* p = pool + (size_t)seg[i] * 32;
  const float* xr = x + (size_t)i * 32;
#pragma unroll
  for (int c = 0; c < UNITS; ++c) atomicAdd(&p[c], xr[c]);
}

__global__ void spmm_kernel(const float* __restrict__ bv,
                            const float* __restrict__ pool,
                            const int* __restrict__ row,
                            const int* __restrict__ col,
                            float* __restrict__ bn, int n)
{
  int i = blockIdx.x * blockDim.x + threadIdx.x;
  if (i >= n) return;
  int cc = col[i], rr = row[i];
  float val = bv[cc];
  const float* p = pool + (size_t)cc * 32;
  float* y = bn + (size_t)rr * 32;
#pragma unroll
  for (int c = 0; c < UNITS; ++c) atomicAdd(&y[c], val * p[c]);
}

__global__ void concat_kernel(const float* __restrict__ a,
                              const float* __restrict__ b,
                              float* __restrict__ cat, int n)
{
  int i = blockIdx.x * blockDim.x + threadIdx.x;
  if (i >= n) return;
  float* y = cat + (size_t)i * 64;
  const float* ar = a + (size_t)i * 32;
  const float* br = b + (size_t)i * 32;
#pragma unroll
  for (int c = 0; c < UNITS; ++c) { y[c] = ar[c]; y[UNITS + c] = br[c]; }
#pragma unroll
  for (int c = 60; c < 64; ++c) y[c] = 0.0f;
}

__global__ void batchpool_kernel(const float* __restrict__ x,
                                 const int* __restrict__ batch,
                                 float* __restrict__ psum,
                                 float* __restrict__ pcnt, int n)
{
  int i = blockIdx.x * blockDim.x + threadIdx.x;
  if (i >= n) return;
  int bb = batch[i];
  atomicAdd(&pcnt[bb], 1.0f);
  float* p = psum + (size_t)bb * 32;
  const float* xr = x + (size_t)i * 32;
#pragma unroll
  for (int c = 0; c < UNITS; ++c) atomicAdd(&p[c], xr[c]);
}

__global__ void head_kernel(const float* __restrict__ psum,
                            const float* __restrict__ pcnt,
                            const float* __restrict__ W2,
                            const float* __restrict__ b2,
                            float* __restrict__ out, int nb)
{
  int i = blockIdx.x * blockDim.x + threadIdx.x;
  if (i >= nb) return;
  float cnt = fmaxf(pcnt[i], 1.0f);
  const float* p = psum + (size_t)i * 32;
  float acc = b2[0];
#pragma unroll
  for (int c = 0; c < UNITS; ++c) acc += W2[c] * (p[c] / cnt);
  out[i] = tanhf(acc);
}

// ---------------------------------------------------------------------------
// Host-side plumbing
// ---------------------------------------------------------------------------
struct ConvP {
  const float *bw, *kW, *kb, *qW, *qb, *sW, *sb, *vW, *vb;
};
static ConvP conv_at(void* const* d_in, int i) {
  ConvP p;
  p.bw = (const float*)d_in[i + 0];
  p.kW = (const float*)d_in[i + 1]; p.kb = (const float*)d_in[i + 2];
  p.qW = (const float*)d_in[i + 3]; p.qb = (const float*)d_in[i + 4];
  p.sW = (const float*)d_in[i + 5]; p.sb = (const float*)d_in[i + 6];
  p.vW = (const float*)d_in[i + 7]; p.vb = (const float*)d_in[i + 8];
  return p;
}

// Packed-weight slots: 2048 f16 (B layout) + 32 f32 (bias) each.
struct PackedLin { const _Float16* B; const float* b; };

static void run_conv(const float* xin, int strideIn, int ksteps, int n,
                     const int* esrc, const int* edst, int ne,
                     const float* betaW,
                     PackedLin pq, PackedLin pk, PackedLin pv, PackedLin ps,
                     float* q, float* k, float* v, float* sk,
                     float* agg, float* sc, float* m, float* s,
                     float* out, hipStream_t stream)
{
  const int nbW = n / 128;                 // 8 waves x 16 rows per 256-thr block
  const int nbN = (n + 255) / 256;
  const int nbE = (ne + 255) / 256;
  const float scale = 1.0f / sqrtf((float)UNITS);

  if (ksteps == 1) {
    qkvs_wmma_kernel<1><<<nbW, 256, 0, stream>>>(xin, strideIn,
        pq.B, pq.b, q, pk.B, pk.b, k, pv.B, pv.b, v, ps.B, ps.b, sk, n);
  } else {
    qkvs_wmma_kernel<2><<<nbW, 256, 0, stream>>>(xin, strideIn,
        pq.B, pq.b, q, pk.B, pk.b, k, pv.B, pv.b, v, ps.B, ps.b, sk, n);
  }

  fill_kernel<<<nbN, 256, 0, stream>>>(m, n, -INFINITY);
  fill_kernel<<<nbN, 256, 0, stream>>>(s, n, 0.0f);
  fill_kernel<<<(long(n) * 32 + 255) / 256, 256, 0, stream>>>(agg, (long)n * 32, 0.0f);

  score_kernel<<<nbE, 256, 0, stream>>>(q, k, esrc, edst, ne, scale, sc, m);
  fixm_kernel<<<nbN, 256, 0, stream>>>(m, n);
  exp_kernel<<<nbE, 256, 0, stream>>>(sc, m, s, edst, ne);
  agg_kernel<<<nbE, 256, 0, stream>>>(sc, s, v, esrc, edst, ne, agg);
  gate_kernel<<<nbN, 256, 0, stream>>>(agg, sk, betaW, out, n, /*relu=*/1);
}

// d_in layout: top-level dict in insertion order; the nested `params` pytree
// is flattened leaf-wise with sorted dict keys at each level
// ('W' < 'b'; convs: beta_W,k,q,skip,v; params: b1,final1,final2,g1,g2,g3,init):
//  0 graph_features  1 income  2 bonus_values_normed
//  3..11  b1     12 final1.W 13 final1.b  14 final2.W 15 final2.b
// 16..24  g1     25..33 g2   34..42 g3    43 init.W   44 init.b
// 45 batch  46 graph_edges  47 bonus_nodes  48 bonus_edges  49 bonus_batch
// 50 bmap_row  51 bmap_col
extern "C" void kernel_launch(void* const* d_in, const int* in_sizes, int n_in,
                              void* d_out, int out_size, void* d_ws, size_t ws_size,
                              hipStream_t stream)
{
  (void)in_sizes; (void)n_in; (void)out_size; (void)ws_size;

  const float* gf  = (const float*)d_in[0];
  const float* inc = (const float*)d_in[1];
  const float* bv  = (const float*)d_in[2];
  const ConvP  b1  = conv_at(d_in, 3);
  const float* f1W = (const float*)d_in[12];
  const float* f1b = (const float*)d_in[13];
  const float* f2W = (const float*)d_in[14];
  const float* f2b = (const float*)d_in[15];
  const ConvP  g1  = conv_at(d_in, 16);
  const ConvP  g2  = conv_at(d_in, 25);
  const ConvP  g3  = conv_at(d_in, 34);
  const float* iW  = (const float*)d_in[43];
  const float* ib  = (const float*)d_in[44];
  const int* batch = (const int*)d_in[45];
  const int* ge    = (const int*)d_in[46];
  const int* bnod  = (const int*)d_in[47];
  const int* be    = (const int*)d_in[48];
  const int* bbat  = (const int*)d_in[49];
  const int* brow  = (const int*)d_in[50];
  const int* bcol  = (const int*)d_in[51];
  float* out = (float*)d_out;

  // workspace carve-up (~185 MB of f32)
  float* W = (float*)d_ws;
  size_t o = 0;
  auto alloc = [&](size_t nf) { float* p = W + o; o += nf; return p; };
  float* hA    = alloc((size_t)NN * 32);
  float* hB    = alloc((size_t)NN * 32);
  float* q     = alloc((size_t)NN * 32);
  float* k     = alloc((size_t)NN * 32);
  float* v     = alloc((size_t)NN * 32);
  float* sk    = alloc((size_t)NN * 32);
  float* agg   = alloc((size_t)NN * 32);
  float* cat   = alloc((size_t)NN * 64);
  float* sc    = alloc((size_t)EE);
  float* m     = alloc((size_t)NN);
  float* s     = alloc((size_t)NN);
  float* bx    = alloc((size_t)NBN * 32);
  float* bpool = alloc((size_t)NBB * 32);
  float* bn    = alloc((size_t)NN * 32);
  float* psum  = alloc((size_t)BBATCH * 32);
  float* pcnt  = alloc((size_t)BBATCH);
  // packed-weight pool: 17 slots x (2048 f16 = 1024 f32) + 17 x 32 f32 bias
  _Float16* wpack = (_Float16*)alloc((size_t)17 * 1024);
  float*    bpad  = alloc((size_t)17 * 32);

  const int* esrc = ge;            // graph_edges[0]
  const int* edst = ge + EE;       // graph_edges[1]
  const int* bsrc = be;            // bonus_edges[0]
  const int* bdst = be + EBN;      // bonus_edges[1]

  // --- pack all 17 linear layers into WMMA B-layout f16 + padded bias ---
  const float* Wmat[17] = { g1.qW, g1.kW, g1.vW, g1.sW,
                            g2.qW, g2.kW, g2.vW, g2.sW,
                            b1.qW, b1.kW, b1.vW, b1.sW,
                            g3.qW, g3.kW, g3.vW, g3.sW,
                            f1W };
  const float* Bvec[17] = { g1.qb, g1.kb, g1.vb, g1.sb,
                            g2.qb, g2.kb, g2.vb, g2.sb,
                            b1.qb, b1.kb, b1.vb, b1.sb,
                            g3.qb, g3.kb, g3.vb, g3.sb,
                            f1b };
  const int Kdim[17] = { 30,30,30,30, 30,30,30,30, 30,30,30,30, 60,60,60,60, 30 };
  PackedLin pl[17];
  for (int i = 0; i < 17; ++i) {
    int ks = (Kdim[i] + 31) / 32;
    int total = ks * 1024;
    _Float16* slotB = wpack + (size_t)i * 2048;
    float*    slotb = bpad + (size_t)i * 32;
    pack_w_kernel<<<(total + 255) / 256, 256, 0, stream>>>(Wmat[i], Kdim[i], ks, slotB);
    pack_b_kernel<<<1, 32, 0, stream>>>(Bvec[i], slotb);
    pl[i].B = slotB; pl[i].b = slotb;
  }

  // 1) init: [N,7] -> relu -> hA
  init_kernel<<<(NN + 255) / 256, 256, 0, stream>>>(gf, inc, batch, iW, ib, hA, NN);

  // 2) g1, g2 on the big graph (ping-pong hA <-> hB)
  run_conv(hA, 32, 1, NN, esrc, edst, EE, g1.bw, pl[0], pl[1], pl[2], pl[3],
           q, k, v, sk, agg, sc, m, s, hB, stream);
  run_conv(hB, 32, 1, NN, esrc, edst, EE, g2.bw, pl[4], pl[5], pl[6], pl[7],
           q, k, v, sk, agg, sc, m, s, hA, stream);

  // 3) bonus sub-graph: gather -> b1 conv -> segment pool -> spmm back to nodes
  gather_kernel<<<(NBN + 255) / 256, 256, 0, stream>>>(hA, bnod, bx, NBN);
  run_conv(bx, 32, 1, NBN, bsrc, bdst, EBN, b1.bw, pl[8], pl[9], pl[10], pl[11],
           q, k, v, sk, agg, sc, m, s, bx, stream);
  fill_kernel<<<((long)NBB * 32 + 255) / 256, 256, 0, stream>>>(bpool, (long)NBB * 32, 0.0f);
  segpool_kernel<<<(NBN + 255) / 256, 256, 0, stream>>>(bx, bbat, bpool, NBN);
  fill_kernel<<<((long)NN * 32 + 255) / 256, 256, 0, stream>>>(bn, (long)NN * 32, 0.0f);
  spmm_kernel<<<(NBN + 255) / 256, 256, 0, stream>>>(bv, bpool, brow, bcol, bn, NBN);

  // 4) concat [hA | bn] -> [N,64] (valid 60), then g3 -> hB
  concat_kernel<<<(NN + 255) / 256, 256, 0, stream>>>(hA, bn, cat, NN);
  run_conv(cat, 64, 2, NN, esrc, edst, EE, g3.bw, pl[12], pl[13], pl[14], pl[15],
           q, k, v, sk, agg, sc, m, s, hB, stream);

  // 5) final1 (relu) -> q buffer; mean pool per graph; head tanh
  linear_wmma_kernel<1><<<NN / 128, 256, 0, stream>>>(hB, 32, pl[16].B, pl[16].b, q, NN, 1);
  fill_kernel<<<((long)BBATCH * 32 + 255) / 256, 256, 0, stream>>>(psum, (long)BBATCH * 32, 0.0f);
  fill_kernel<<<(BBATCH + 255) / 256, 256, 0, stream>>>(pcnt, (long)BBATCH, 0.0f);
  batchpool_kernel<<<(NN + 255) / 256, 256, 0, stream>>>(q, batch, psum, pcnt, NN);
  head_kernel<<<(BBATCH + 255) / 256, 256, 0, stream>>>(psum, pcnt, f2W, f2b, out, BBATCH);
}
